// RWKVBlock_21844203667890
// MI455X (gfx1250) — compile-verified, matrix operand order checked by reading, not run on hardware
//
#include <hip/hip_runtime.h>

#define DEV __device__ __forceinline__

// ---- CDNA5 WMMA vector types (gfx1250, wave32) ----
typedef __attribute__((ext_vector_type(16))) __bf16 v16bf;
typedef __attribute__((ext_vector_type(8)))  __bf16 v8bf;
typedef __attribute__((ext_vector_type(8)))  float  v8f;

union FragAB { v16bf v; v8bf h[2]; };

DEV unsigned short f2bf(float f) {
  unsigned u = __float_as_uint(f);
  unsigned r = u + 0x7FFFu + ((u >> 16) & 1u);   // round-to-nearest-even
  return (unsigned short)(r >> 16);
}

// LDS aperture is 2^32-aligned (ISA 10.2: isLDS checks addr[63:32] only), so
// the low 32 bits of a flat address into __shared__ memory are the LDS byte
// offset that GLOBAL_LOAD_ASYNC_TO_LDS_* expects in its VDST VGPR.
DEV unsigned lds_addr(const void* p) {
  return (unsigned)(unsigned long long)p;
}

// CDNA5 async cache->LDS copy (ISA 15.18.3 op 98, tracked by ASYNCcnt).
DEV void async_load_b128(unsigned lds_off, const void* gaddr) {
  asm volatile("global_load_async_to_lds_b128 %0, %1, off"
               :: "v"(lds_off), "v"(gaddr)
               : "memory");
}
DEV void wait_async0() {
  asm volatile("s_wait_asynccnt 0x0" ::: "memory");
}

// ---------------------------------------------------------------------------
// fp32 -> bf16 conversion (weights / activations), grid-stride
// ---------------------------------------------------------------------------
__global__ void f32_to_bf16_kernel(const float* __restrict__ src,
                                   unsigned short* __restrict__ dst,
                                   size_t n) {
  size_t i = (size_t)blockIdx.x * blockDim.x + threadIdx.x;
  size_t stride = (size_t)gridDim.x * blockDim.x;
  for (; i < n; i += stride) dst[i] = f2bf(src[i]);
}

// ---------------------------------------------------------------------------
// Block reduction helper (256 threads)
// ---------------------------------------------------------------------------
DEV float blockReduce256(float v, float* sm) {
  const int tid = threadIdx.x;
  sm[tid] = v;
  __syncthreads();
#pragma unroll
  for (int off = 128; off > 0; off >>= 1) {
    if (tid < off) sm[tid] += sm[tid + off];
    __syncthreads();
  }
  float r = sm[0];
  __syncthreads();
  return r;
}

// ---------------------------------------------------------------------------
// LayerNorm + time_shift fused, bf16 outputs.
// One block per (b,t) row of C=1024; 256 threads, 4 channels per thread.
// h = LN(x[t]); hp = LN(x[t-1]) (zeros at t==0); out_j = h*mix_j + hp*(1-mix_j)
// ---------------------------------------------------------------------------
template <int NOUT>
__global__ __launch_bounds__(256)
void ln_shift_kernel(const float* __restrict__ x,
                     const float* __restrict__ gamma,
                     const float* __restrict__ beta,
                     const float* __restrict__ mix0,
                     const float* __restrict__ mix1,
                     const float* __restrict__ mix2,
                     unsigned short* __restrict__ o0,
                     unsigned short* __restrict__ o1,
                     unsigned short* __restrict__ o2,
                     int T, int C) {
  __shared__ float sm[256];
  const int row = blockIdx.x;           // b*T + t
  const int t = row % T;
  const float* xc = x + (size_t)row * C;
  const float* xp = xc - C;             // valid iff t > 0

  float cur[4], prv[4];
  float s = 0.f, s2 = 0.f, sp = 0.f, sp2 = 0.f;
#pragma unroll
  for (int i = 0; i < 4; ++i) {
    int c = threadIdx.x + i * 256;
    float a = xc[c];
    cur[i] = a; s += a; s2 += a * a;
    float p = (t > 0) ? xp[c] : 0.0f;
    prv[i] = p; sp += p; sp2 += p * p;
  }
  s   = blockReduce256(s,   sm);
  s2  = blockReduce256(s2,  sm);
  sp  = blockReduce256(sp,  sm);
  sp2 = blockReduce256(sp2, sm);

  const float inv = 1.0f / (float)C;
  float mu  = s * inv;
  float rs  = rsqrtf(s2 * inv - mu * mu + 1e-5f);
  float mup = sp * inv;
  float rsp = rsqrtf(sp2 * inv - mup * mup + 1e-5f);

  unsigned short* outs[3] = {o0, o1, o2};
  const float* mixes[3] = {mix0, mix1, mix2};

#pragma unroll
  for (int i = 0; i < 4; ++i) {
    int c = threadIdx.x + i * 256;
    float g = gamma[c], bt = beta[c];
    float h  = (cur[i] - mu) * rs * g + bt;
    float hp = (t > 0) ? (prv[i] - mup) * rsp * g + bt : 0.0f;
    size_t idx = (size_t)row * C + c;
#pragma unroll
    for (int j = 0; j < NOUT; ++j) {
      float mk = mixes[j][c];
      outs[j][idx] = f2bf(h * mk + hp * (1.0f - mk));
    }
  }
}

// ---------------------------------------------------------------------------
// WMMA GEMM: out[M,N] = A[M,K] (bf16, row-major) x W[N,K]^T (bf16, row-major)
//
// Block tile 128(M) x 128(N), 256 threads = 8 waves in a 4(M) x 2(N) grid;
// each wave owns a 32x64 output tile: 2 A-frags x 4 B-frags -> 8 WMMAs with
// 12 ds_load_b128 per K-step (vs 18 for a 16x128 wave tile).
//
// K-step 32, LDS double-buffered, staged with GLOBAL_LOAD_ASYNC_TO_LDS_B128
// (ASYNCcnt): asyncs for step kt+1 are issued before computing on buffer kt,
// then a single s_wait_asynccnt 0 + barrier per step. Tile rows padded to 40
// bf16 elements so 16 fragment rows hit 16 distinct LDS banks.
//
// bf16 A fragment (16x32), lane layout per CDNA5 ISA 7.12.2:
//   lane L: row M = L%16; K-halves = (L<16) ? {0..7,16..23} : {8..15,24..31}
// B fragment (32x16) is the mirror with N = L%16.  C/D: VGPR r -> M = r+hi*8,
// N = lane%16.
//
// ACT: 0 none, 1 sigmoid, 2 relu^2.  OUTBF: write bf16 instead of f32.
// FUSE: out = addsrc + mulsrc * acc  (final residual+gate epilogue).
// ---------------------------------------------------------------------------
#define LDS_STRIDE 40   // 32 payload + 8 pad bf16 elements per row

template <int ACT, int OUTBF, int FUSE>
__global__ __launch_bounds__(256)
void wmma_gemm_kernel(const unsigned short* __restrict__ A,
                      const unsigned short* __restrict__ W,
                      float* __restrict__ outF,
                      unsigned short* __restrict__ outH,
                      const float* __restrict__ addsrc,
                      const float* __restrict__ mulsrc,
                      int M, int N, int K) {
  // [buf][0]=A tile, [buf][1]=B tile; 128 rows x 32 (+8 pad) bf16
  __shared__ unsigned short smem[2][2][128 * LDS_STRIDE];

  const int tid   = threadIdx.x;
  const int wave  = tid >> 5;
  const int lane  = tid & 31;
  const int laneN = lane & 15;
  const int hi    = lane >> 4;                 // K-half selector
  const int waveM = wave & 3;                  // 4 M-subtiles of 32 rows
  const int waveN = wave >> 2;                 // 2 N-subtiles of 64 cols
  const int mrow0 = blockIdx.x * 128;
  const int nbase = blockIdx.y * 128;
  const int mbase = mrow0 + waveM * 32;
  const int nwav0 = waveN * 64;                // within block N-tile

  // chunk mapping for async staging: 512 chunks of 8 bf16 per tile,
  // each thread issues 2 A-chunks + 2 B-chunks per K-step.
  const int c0row = (tid + 0)   >> 2;
  const int c0k   = ((tid + 0)  & 3) * 8;
  const int c1row = (tid + 256) >> 2;
  const int c1k   = ((tid + 256) & 3) * 8;

  const size_t Kz = (size_t)K;

  v8f acc[2][4] = {};

  auto issue_tile = [&](int buf, int k0) {
    async_load_b128(lds_addr(&smem[buf][0][c0row * LDS_STRIDE + c0k]),
                    A + (size_t)(mrow0 + c0row) * Kz + k0 + c0k);
    async_load_b128(lds_addr(&smem[buf][0][c1row * LDS_STRIDE + c1k]),
                    A + (size_t)(mrow0 + c1row) * Kz + k0 + c1k);
    async_load_b128(lds_addr(&smem[buf][1][c0row * LDS_STRIDE + c0k]),
                    W + (size_t)(nbase + c0row) * Kz + k0 + c0k);
    async_load_b128(lds_addr(&smem[buf][1][c1row * LDS_STRIDE + c1k]),
                    W + (size_t)(nbase + c1row) * Kz + k0 + c1k);
  };

  // ---- prologue: stage K-step 0 ----
  issue_tile(0, 0);
  wait_async0();
  __syncthreads();

  const int nk = K >> 5;                       // K / 32
  const int aOff0 = (waveM * 32 + laneN) * LDS_STRIDE + hi * 8;
  const int bOff0 = (nwav0 + laneN) * LDS_STRIDE + hi * 8;

  for (int kt = 0; kt < nk; ++kt) {
    const int cur = kt & 1;
    const bool more = (kt + 1) < nk;

    // issue async copies for next K-step into the other buffer
    // (safe: that buffer was last read in step kt-1, ordered by the barrier)
    if (more) issue_tile(cur ^ 1, (kt + 1) << 5);

    // ---- compute on LDS buffer `cur` ----
    const unsigned short* Abuf = smem[cur][0];
    const unsigned short* Bbuf = smem[cur][1];

    FragAB a[2], b[4];
#pragma unroll
    for (int mi = 0; mi < 2; ++mi) {
      const int off = aOff0 + mi * 16 * LDS_STRIDE;
      a[mi].h[0] = *reinterpret_cast<const v8bf*>(Abuf + off);
      a[mi].h[1] = *reinterpret_cast<const v8bf*>(Abuf + off + 16);
    }
#pragma unroll
    for (int ni = 0; ni < 4; ++ni) {
      const int off = bOff0 + ni * 16 * LDS_STRIDE;
      b[ni].h[0] = *reinterpret_cast<const v8bf*>(Bbuf + off);
      b[ni].h[1] = *reinterpret_cast<const v8bf*>(Bbuf + off + 16);
    }
#pragma unroll
    for (int mi = 0; mi < 2; ++mi)
#pragma unroll
      for (int ni = 0; ni < 4; ++ni)
        acc[mi][ni] = __builtin_amdgcn_wmma_f32_16x16x32_bf16(
            false, a[mi].v, false, b[ni].v, (short)0, acc[mi][ni], false, false);

    // ---- retire next tile's asyncs, flip buffers ----
    if (more) {
      wait_async0();
      __syncthreads();
    }
  }

  // C/D layout: VGPR r -> M = mbase + mi*16 + r + hi*8, N = nwav0 + ni*16 + laneN
#pragma unroll
  for (int mi = 0; mi < 2; ++mi) {
#pragma unroll
    for (int ni = 0; ni < 4; ++ni) {
#pragma unroll
      for (int r = 0; r < 8; ++r) {
        float v = acc[mi][ni][r];
        if (ACT == 1) v = 1.0f / (1.0f + __expf(-v));
        else if (ACT == 2) v = (v > 0.0f) ? v * v : 0.0f;
        const int m = mbase + mi * 16 + r + hi * 8;
        const int n = nbase + nwav0 + ni * 16 + laneN;
        const size_t idx = (size_t)m * N + n;
        float o = v;
        if (FUSE) o = addsrc[idx] + mulsrc[idx] * v;
        if (OUTBF) outH[idx] = f2bf(o);
        else       outF[idx] = o;
      }
    }
  }
}

// ---------------------------------------------------------------------------
// WKV recurrence: one thread per (b,c) channel, serial over T.
// Writes x1 = x + r * wkv_out and final state (B,C,3) = {aa, bb, pp}.
// ---------------------------------------------------------------------------
__global__ __launch_bounds__(256)
void wkv_kernel(const float* __restrict__ x,
                const float* __restrict__ k,
                const float* __restrict__ v,
                const float* __restrict__ r,
                const float* __restrict__ time_decay,
                const float* __restrict__ time_first,
                float* __restrict__ x1,
                float* __restrict__ state,   // (B, C, 3)
                int Bn, int T, int C) {
  int tid = blockIdx.x * blockDim.x + threadIdx.x;
  if (tid >= Bn * C) return;
  int b = tid / C, c = tid % C;

  const float w = -__expf(time_decay[c]);
  const float u = time_first[c];
  const float NINF = -__builtin_inff();
  float aa = 0.0f, bb = NINF, pp = NINF;

  size_t idx = (size_t)b * T * C + c;
  for (int t = 0; t < T; ++t, idx += C) {
    float kk = k[idx], vv = v[idx];
    float ww = u + kk;
    float p  = fmaxf(pp, ww);
    float e1 = __expf(pp - p), e2 = __expf(ww - p);
    float out = (e1 * aa + e2 * vv) / (e1 + e2 + 1e-8f);
    x1[idx] = x[idx] + r[idx] * out;

    float bbn = w + pp;                       // decayed value, stored in bb only
    float p2  = fmaxf(pp, kk);
    float e1b = __expf(pp - p2), e2b = __expf(kk - p2);
    aa = e1b * aa + e2b * vv;
    pp = p2 + __logf(e1b + e2b + 1e-8f);
    bb = bbn;
  }
  float* st = state + (size_t)(b * C + c) * 3;
  st[0] = aa; st[1] = bb; st[2] = pp;
}

// ---------------------------------------------------------------------------
// Host-side orchestration
// ---------------------------------------------------------------------------
extern "C" void kernel_launch(void* const* d_in, const int* in_sizes, int n_in,
                              void* d_out, int out_size, void* d_ws, size_t ws_size,
                              hipStream_t stream) {
  (void)in_sizes; (void)n_in; (void)out_size; (void)ws_size;
  const int B = 8, T = 2048, C = 1024, C4 = 4096;
  const int MT = B * T;                 // 16384 rows

  const float* x          = (const float*)d_in[0];
  const float* time_decay = (const float*)d_in[1];
  const float* time_first = (const float*)d_in[2];
  const float* Wk_t = (const float*)d_in[3];
  const float* Wv_t = (const float*)d_in[4];
  const float* Wr_t = (const float*)d_in[5];
  const float* Wk_c = (const float*)d_in[6];   // (4C, C)
  const float* Wv_c = (const float*)d_in[7];   // (C, 4C)
  const float* Wr_c = (const float*)d_in[8];
  const float* ln1_g = (const float*)d_in[9];
  const float* ln1_b = (const float*)d_in[10];
  const float* ln2_g = (const float*)d_in[11];
  const float* ln2_b = (const float*)d_in[12];
  const float* mix_k = (const float*)d_in[13];
  const float* mix_v = (const float*)d_in[14];
  const float* mix_r = (const float*)d_in[15];
  const float* cmix_k = (const float*)d_in[16];
  const float* cmix_r = (const float*)d_in[17];

  float* xout  = (float*)d_out;              // (B,T,C)
  float* state = xout + (size_t)MT * C;      // (B,C,3)

  // ---- carve workspace ----
  char* ws = (char*)d_ws;
  size_t off = 0;
  auto carve = [&](size_t bytes) -> void* {
    void* p = ws + off;
    off = (off + bytes + 255) & ~(size_t)255;
    return p;
  };
  unsigned short* wkt = (unsigned short*)carve((size_t)C * C  * 2);
  unsigned short* wvt = (unsigned short*)carve((size_t)C * C  * 2);
  unsigned short* wrt = (unsigned short*)carve((size_t)C * C  * 2);
  unsigned short* wkc = (unsigned short*)carve((size_t)C4 * C * 2);
  unsigned short* wvc = (unsigned short*)carve((size_t)C * C4 * 2);
  unsigned short* wrc = (unsigned short*)carve((size_t)C * C  * 2);
  unsigned short* xk  = (unsigned short*)carve((size_t)MT * C * 2);
  unsigned short* xv  = (unsigned short*)carve((size_t)MT * C * 2);
  unsigned short* xr  = (unsigned short*)carve((size_t)MT * C * 2);
  float* kf  = (float*)carve((size_t)MT * C * 4);
  float* vf  = (float*)carve((size_t)MT * C * 4);
  float* rf  = (float*)carve((size_t)MT * C * 4);
  float* x1  = (float*)carve((size_t)MT * C * 4);
  unsigned short* xk2 = (unsigned short*)carve((size_t)MT * C * 2);
  unsigned short* xr2 = (unsigned short*)carve((size_t)MT * C * 2);
  unsigned short* kc  = (unsigned short*)carve((size_t)MT * C4 * 2);
  float* rcf = (float*)carve((size_t)MT * C * 4);

  // ---- weights fp32 -> bf16 (L2-resident afterwards) ----
  f32_to_bf16_kernel<<<1024, 256, 0, stream>>>(Wk_t, wkt, (size_t)C * C);
  f32_to_bf16_kernel<<<1024, 256, 0, stream>>>(Wv_t, wvt, (size_t)C * C);
  f32_to_bf16_kernel<<<1024, 256, 0, stream>>>(Wr_t, wrt, (size_t)C * C);
  f32_to_bf16_kernel<<<1024, 256, 0, stream>>>(Wk_c, wkc, (size_t)C4 * C);
  f32_to_bf16_kernel<<<1024, 256, 0, stream>>>(Wv_c, wvc, (size_t)C * C4);
  f32_to_bf16_kernel<<<1024, 256, 0, stream>>>(Wr_c, wrc, (size_t)C * C);

  // ---- time mixing ----
  ln_shift_kernel<3><<<MT, 256, 0, stream>>>(x, ln1_g, ln1_b, mix_k, mix_v, mix_r,
                                             xk, xv, xr, T, C);

  dim3 g1(MT / 128, C / 128);
  dim3 g4(MT / 128, C4 / 128);
  wmma_gemm_kernel<0, 0, 0><<<g1, 256, 0, stream>>>(xk, wkt, kf, nullptr, nullptr, nullptr, MT, C, C);
  wmma_gemm_kernel<0, 0, 0><<<g1, 256, 0, stream>>>(xv, wvt, vf, nullptr, nullptr, nullptr, MT, C, C);
  wmma_gemm_kernel<1, 0, 0><<<g1, 256, 0, stream>>>(xr, wrt, rf, nullptr, nullptr, nullptr, MT, C, C);

  wkv_kernel<<<(B * C + 255) / 256, 256, 0, stream>>>(x, kf, vf, rf, time_decay,
                                                      time_first, x1, state, B, T, C);

  // ---- channel mixing ----
  ln_shift_kernel<2><<<MT, 256, 0, stream>>>(x1, ln2_g, ln2_b, cmix_k, cmix_r, nullptr,
                                             xk2, xr2, nullptr, T, C);

  wmma_gemm_kernel<2, 1, 0><<<g4, 256, 0, stream>>>(xk2, wkc, nullptr, kc, nullptr, nullptr, MT, C4, C);
  wmma_gemm_kernel<1, 0, 0><<<g1, 256, 0, stream>>>(xr2, wrc, rcf, nullptr, nullptr, nullptr, MT, C, C);
  // vv = relu(kc)^2 @ Wv_c^T, fused: out = x1 + rc * vv  -> straight into d_out
  wmma_gemm_kernel<0, 0, 1><<<g1, 256, 0, stream>>>(kc, wvc, xout, nullptr, x1, rcf, MT, C, C4);
}
